// LSTM_69707319214566
// MI455X (gfx1250) — matrix operand order repaired and checked by reading, not compiled, not run on hardware
//
#include <hip/hip_runtime.h>
#include <hip/hip_bf16.h>

typedef _Float16 h4   __attribute__((ext_vector_type(4)));
typedef _Float16 h8   __attribute__((ext_vector_type(8)));
typedef _Float16 h16  __attribute__((ext_vector_type(16)));
typedef float    f8   __attribute__((ext_vector_type(8)));

#define B_    2048
#define T_    128
#define DIN   64
#define U_    128
#define NG    512              // 4*U
#define BM    16               // batch rows per workgroup (256 WGs total)
#define XH_STRIDE 208          // f16 per staged row (192 used + pad; rows 16B-aligned)
#define WFRAG_ELEMS (6*32*32*16)   // 6 k-chunks * 32 n-tiles * 32 lanes * 16 f16 = 98304
#define EPS_  1e-5f

// ---- fast activations on gfx1250 hardware trans pipe -----------------------
__device__ __forceinline__ float fast_rcp(float x) {
#if __has_builtin(__builtin_amdgcn_rcpf)
    return __builtin_amdgcn_rcpf(x);
#else
    return 1.0f / x;
#endif
}
__device__ __forceinline__ float fast_exp2(float x) {
#if __has_builtin(__builtin_amdgcn_exp2f)
    return __builtin_amdgcn_exp2f(x);
#else
    return exp2f(x);
#endif
}
__device__ __forceinline__ float sigm(float x) {
    // 1/(1+2^(-x*log2e)) : v_mul + v_exp + v_add + v_rcp
    return fast_rcp(1.0f + fast_exp2(-1.4426950408889634f * x));
}
__device__ __forceinline__ float fast_tanh(float x) {
#if __has_builtin(__builtin_amdgcn_tanh_f32)
    return __builtin_amdgcn_tanh_f32(x);      // hardware V_TANH_F32
#elif __has_builtin(__builtin_amdgcn_tanhf)
    return __builtin_amdgcn_tanhf(x);
#else
    // tanh(x) = 1 - 2/(2^(2x*log2e)+1)
    return 1.0f - 2.0f * fast_rcp(1.0f + fast_exp2(2.8853900817779268f * x));
#endif
}

// ---------------------------------------------------------------------------
// Kernel 1: pack Wc = [W_ih.T | W_hh.T] (K=192, N=512) into WMMA B-fragment
// order (f16), fold biases. Fragment element (kc,nt,lane,e):
//   n = nt*16 + lane%16 ; k = kc*32 + (lane/16)*8 + e (e<8), kc*32 + 16 + (lane/16)*8 + (e-8) (e>=8)
// Each lane's 16 f16 form one contiguous 32-byte run -> 2x global_load_b128.
// ---------------------------------------------------------------------------
__global__ void pack_weights(const float* __restrict__ Wih,
                             const float* __restrict__ Whh,
                             const float* __restrict__ bih,
                             const float* __restrict__ bhh,
                             _Float16* __restrict__ wf,
                             float* __restrict__ bias) {
    int tid = blockIdx.x * 256 + threadIdx.x;
    if (tid < WFRAG_ELEMS) {
        int e  = tid & 15;
        int l  = (tid >> 4) & 31;
        int nt = (tid >> 9) & 31;
        int kc = tid >> 14;
        int n  = nt * 16 + (l & 15);
        int k  = kc * 32 + (e >> 3) * 16 + (l >> 4) * 8 + (e & 7);
        float v = (k < DIN) ? Wih[n * DIN + k] : Whh[n * U_ + (k - DIN)];
        wf[tid] = (_Float16)v;
    }
    if (tid < NG) bias[tid] = bih[tid] + bhh[tid];
}

// ---------------------------------------------------------------------------
// Kernel 2: fused dual-path LSTM recurrence + temporal mean + BatchNorm.
// grid = (B/16, 2): blockIdx.y==0 -> SNN (input_feature_st),
//                   blockIdx.y==1 -> ANN (input_features_sc).
// 256 threads = 8 wave32. Wave w owns units u in [16w,16w+16) for all 4 gates
// (n-tiles {w, w+8, w+16, w+24}) over one 16-row m-tile, so i/f/g/o for a
// given (m,u) live in the same lane/VGPR slot: c/mem/pool state stays in
// registers with zero cross-lane traffic. Weight B-fragments are loop-
// invariant -> compiler keeps them VGPR-resident across all 128 steps.
// ---------------------------------------------------------------------------
__global__ __launch_bounds__(256)
void lstm_dual(const float* __restrict__ x_st,
               const float* __restrict__ x_sc,
               const _Float16* __restrict__ wf,
               const float* __restrict__ bias,
               const float* __restrict__ gamma,
               const float* __restrict__ beta,
               const float* __restrict__ mean,
               const float* __restrict__ var,
               float* __restrict__ out) {
    __shared__ _Float16 xh[BM * XH_STRIDE];   // [x_t(64) | h(128) | pad] per row

    const int  tid  = threadIdx.x;
    const int  lane = tid & 31;
    const int  w    = tid >> 5;                 // wave 0..7
    const int  b0   = blockIdx.x * BM;
    const bool snn  = (blockIdx.y == 0);
    const float* x  = snn ? x_st : x_sc;

    const int u     = (w << 4) + (lane & 15);   // unit 0..127 (this lane's column)
    const int mrow  = (lane >> 4) * 8;          // +r -> batch row inside tile
    const int khalf = (lane >> 4) * 8;          // A-fragment K half select

    const float bi = bias[u];
    const float bf = bias[U_ + u];
    const float bg = bias[2 * U_ + u];
    const float bo = bias[3 * U_ + u];

    // zero the h region of the staging buffer (h_0 = 0)
    {
        int m  = tid >> 4;                      // 0..15
        int ub = (tid & 15) << 3;               // 0,8,..,120
        h8 z;
        #pragma unroll
        for (int i = 0; i < 8; ++i) z[i] = (_Float16)0.0f;
        *(h8*)&xh[m * XH_STRIDE + DIN + ub] = z;
    }

    f8    acc[4];
    float c[8], memv[8], pool[8];
    #pragma unroll
    for (int r = 0; r < 8; ++r) { c[r] = 0.f; memv[r] = 0.f; pool[r] = 0.f; }

    const int xrow = tid >> 4;            // 0..15
    const int xcol = (tid & 15) << 2;     // 0,4,..,60
    const float* xbase = x + (size_t)(b0 + xrow) * T_ * DIN + xcol;

    for (int t = 0; t < T_; ++t) {
        // ---- stage x_t tile into LDS (f32 -> f16) ----
        float4 v0 = *(const float4*)(xbase + t * DIN);
        __builtin_prefetch(xbase + (t + 1) * DIN, 0, 0);   // global_prefetch next step
        h4 hx;
        hx[0] = (_Float16)v0.x; hx[1] = (_Float16)v0.y;
        hx[2] = (_Float16)v0.z; hx[3] = (_Float16)v0.w;
        *(h4*)&xh[xrow * XH_STRIDE + xcol] = hx;
        __syncthreads();   // x_t and h_{t-1} visible to all waves

        // ---- gates = [x|h] @ Wc via WMMA, K=192 in six 32-chunks ----
        #pragma unroll
        for (int g = 0; g < 4; ++g)
            #pragma unroll
            for (int r = 0; r < 8; ++r) acc[g][r] = 0.0f;

        #pragma unroll
        for (int kc = 0; kc < 6; ++kc) {
            // A fragment (ISA 16-bit A layout: M=lane%16, K half by lane/16)
            const _Float16* p0 = &xh[(lane & 15) * XH_STRIDE + kc * 32 + khalf];
            h8 lo = *(const h8*)p0, hi = *(const h8*)(p0 + 16);
            h16 a;
            #pragma unroll
            for (int e = 0; e < 8; ++e) { a[e] = lo[e]; a[8 + e] = hi[e]; }
            #pragma unroll
            for (int g = 0; g < 4; ++g) {
                const _Float16* bp = wf + ((size_t)((kc * 32 + (g * 8 + w)) * 32 + lane) << 4);
                h16 bfrag = *(const h16*)bp;   // loop-invariant -> register-resident
                acc[g] = __builtin_amdgcn_wmma_f32_16x16x32_f16(
                    false, a, false, bfrag, (short)0, acc[g], false, false);
            }
        }
        __syncthreads();   // all LDS reads done before h is overwritten

        // ---- per-(m,u) recurrence update, all in registers ----
        #pragma unroll
        for (int r = 0; r < 8; ++r) {
            float gi = sigm(acc[0][r] + bi);
            float gf = sigm(acc[1][r] + bf);
            float gg = fast_tanh(acc[2][r] + bg);
            float go = sigm(acc[3][r] + bo);
            float cc = gf * c[r] + gi * gg;
            c[r] = cc;
            float hv = go * fast_tanh(cc);
            float hw;
            if (snn) {                       // integrate-and-fire, soft reset
                float m2  = memv[r] + hv;
                float spk = (m2 >= 1.0f) ? 1.0f : 0.0f;
                memv[r] = m2 - spk;
                pool[r] += spk;
                hw = spk;                    // carry h = spike
            } else {
                pool[r] += hv;
                hw = hv;
            }
            int m = mrow + r;
            xh[m * XH_STRIDE + DIN + u] = (_Float16)hw;
        }
    }

    // ---- temporal mean + BatchNorm (eval) + store ----
    const float invT = 1.0f / (float)T_;
    const float ga = gamma[u], be = beta[u], mu = mean[u];
    const float iv = rsqrtf(var[u] + EPS_);
    const size_t obase = (size_t)blockIdx.y * B_ * U_;
    #pragma unroll
    for (int r = 0; r < 8; ++r) {
        int m = mrow + r;
        float p = pool[r] * invT;
        out[obase + (size_t)(b0 + m) * U_ + u] = (p - mu) * iv * ga + be;
    }
}

// ---------------------------------------------------------------------------
extern "C" void kernel_launch(void* const* d_in, const int* in_sizes, int n_in,
                              void* d_out, int out_size, void* d_ws, size_t ws_size,
                              hipStream_t stream) {
    const float* x_st  = (const float*)d_in[0];
    const float* x_sc  = (const float*)d_in[1];
    const float* Wih   = (const float*)d_in[2];
    const float* Whh   = (const float*)d_in[3];
    const float* bih   = (const float*)d_in[4];
    const float* bhh   = (const float*)d_in[5];
    const float* gamma = (const float*)d_in[6];
    const float* beta  = (const float*)d_in[7];
    const float* mean  = (const float*)d_in[8];
    const float* var   = (const float*)d_in[9];

    _Float16* wf   = (_Float16*)d_ws;
    float*    bias = (float*)((char*)d_ws + (size_t)WFRAG_ELEMS * sizeof(_Float16));

    pack_weights<<<(WFRAG_ELEMS + 255) / 256, 256, 0, stream>>>(Wih, Whh, bih, bhh, wf, bias);

    dim3 grid(B_ / BM, 2);
    lstm_dual<<<grid, 256, 0, stream>>>(x_st, x_sc, wf, bias, gamma, beta, mean, var,
                                        (float*)d_out);
}
